// InsideAlgorithm_5952824672518
// MI455X (gfx1250) — matrix operand order repaired and checked by reading, not compiled, not run on hardware
//
#include <hip/hip_runtime.h>
#include <hip/hip_bf16.h>

typedef __attribute__((ext_vector_type(16))) __bf16 v16bf;
typedef __attribute__((ext_vector_type(8)))  __bf16 v8bf;
typedef __attribute__((ext_vector_type(8)))  float  v8f;

#define LSEQ 32
#define NSYM 48
#define NN   2304   // NSYM*NSYM

// LDS layout (bytes) -- fixed for every step
#define EOFF_L 0        // eLT: 16 spans * 48 rows * 32 bf16 = 49152 B
#define EOFF_R 49152    // eRT: same
#define GOFF   98304    // gT : 16 cols * 2304 bf16 = 73728 B
#define REDOFF 172032   // red: 6 waves * 8 * 32 f32 = 6144 B
#define MLAOFF 178176   // mLa: 256 f32
#define MRAOFF 179200   // mRa: 256 f32
#define MSAOFF 180224   // mSa: 256 f32
#define MPAOFF 181248   // mPa: 32 f32
#define LDSBYTES 181376

__device__ __forceinline__ __bf16 f32_to_bf16(float f) {
  unsigned u = __builtin_bit_cast(unsigned, f);
  unsigned r = (u + 0x7FFFu + ((u >> 16) & 1u)) >> 16;
  return __builtin_bit_cast(__bf16, (unsigned short)r);
}
__device__ __forceinline__ __bf16 bf16_zero() {
  return __builtin_bit_cast(__bf16, (unsigned short)0);
}

// ---- W2[a, bR*48+bL] = clip(binary[a,bL,bR]) / max;  mW[a] = log(max) ----
__global__ void prep_w_kernel(const float* __restrict__ binary,
                              __bf16* __restrict__ w2, float* __restrict__ mW) {
  int a = blockIdx.x;
  const float* src = binary + (size_t)a * NN;
  __shared__ float red[256];
  float mx = 1e-30f;
  for (int c = threadIdx.x; c < NN; c += 256) mx = fmaxf(mx, fmaxf(src[c], 1e-30f));
  red[threadIdx.x] = mx; __syncthreads();
  for (int o = 128; o > 0; o >>= 1) {
    if (threadIdx.x < o) red[threadIdx.x] = fmaxf(red[threadIdx.x], red[threadIdx.x + o]);
    __syncthreads();
  }
  float mv = red[0];
  if (threadIdx.x == 0) mW[a] = logf(mv);
  float inv = 1.0f / mv;
  for (int c = threadIdx.x; c < NN; c += 256) {
    int bL = c / NSYM, bR = c % NSYM;
    w2[(size_t)a * NN + bR * NSYM + bL] = f32_to_bf16(fmaxf(src[c], 1e-30f) * inv);
  }
}

// ---- chart[b][i][i+1][n] = log(clip(lexical[n, tokens[b,i]])) ----
__global__ void diag_kernel(const int* __restrict__ tokens,
                            const float* __restrict__ lexical,
                            float* __restrict__ chart) {
  int idx = blockIdx.x * blockDim.x + threadIdx.x;   // B*L*N
  if (idx >= 32 * LSEQ * NSYM) return;
  int n = idx % NSYM; int bi = idx / NSYM;
  int i = bi % LSEQ;  int b = bi / LSEQ;
  int tok = tokens[b * LSEQ + i];
  float v = logf(fmaxf(lexical[(size_t)n * 32000 + tok], 1e-30f));
  chart[((((size_t)b * LSEQ + i) * (LSEQ + 1)) + (i + 1)) * NSYM + n] = v;
}

// ---- one width-step: all spans of width s, one workgroup per batch ----
__global__ __launch_bounds__(256, 1)
void span_step_kernel(float* __restrict__ chart,
                      const __bf16* __restrict__ w2,
                      const float* __restrict__ mW, int s) {
  extern __shared__ char smem[];
  const int b   = blockIdx.x;
  const int Km  = s - 1;                 // split contraction, 1..31
  const int pos = LSEQ - s + 1;          // spans this step, 1..31
  const int P   = pos * Km;              // <= 256
  const int tid = threadIdx.x;

  __bf16* eLT  = (__bf16*)(smem + EOFF_L);   // [cc][n][m32]
  __bf16* eRT  = (__bf16*)(smem + EOFF_R);   // [cc][n][m32]
  __bf16* gT   = (__bf16*)(smem + GOFF);     // [cc][c], c = bR*48+bL
  float*  redp = (float*) (smem + REDOFF);   // [wave][8][32]
  float*  mLa  = (float*) (smem + MLAOFF);
  float*  mRa  = (float*) (smem + MRAOFF);
  float*  mSa  = (float*) (smem + MSAOFF);
  float*  mPa  = (float*) (smem + MPAOFF);

  // ---- Phase A1: per (span,split) row maxes over symbols ----
  if (tid < P) {
    int i = tid / Km, m = tid % Km;
    int k = i + 1 + m, j = i + s;
    const float* lrow = chart + (((size_t)b * LSEQ + i) * (LSEQ + 1) + k) * NSYM;
    const float* rrow = chart + (((size_t)b * LSEQ + k) * (LSEQ + 1) + j) * NSYM;
    float ml = -3.0e38f, mr = -3.0e38f;
    for (int n = 0; n < NSYM; ++n) { ml = fmaxf(ml, lrow[n]); mr = fmaxf(mr, rrow[n]); }
    mLa[tid] = ml; mRa[tid] = mr; mSa[tid] = ml + mr;
  }
  __syncthreads();
  if (tid < pos) {
    float mp = -3.0e38f;
    for (int m = 0; m < Km; ++m) mp = fmaxf(mp, mSa[tid * Km + m]);
    mPa[tid] = mp;
  }

  const int wave = tid >> 5, lane = tid & 31;
  const int r = lane & 15, hi = lane >> 4;

  // ---- process spans in chunks of 16 ----
  for (int i0 = 0; i0 < pos; i0 += 16) {
    const int cnt = (pos - i0 < 16) ? (pos - i0) : 16;
    __syncthreads();   // previous chunk's consumers done; mPa ready on first pass

    // Phase A2: max-shifted exponentials, zero-padded to m=32, bf16, transposed
    for (int row = tid; row < cnt * NSYM; row += 256) {
      int cc = row / NSYM, n = row % NSYM;
      int i = i0 + cc, j = i + s;
      __bf16 bufL[32], bufR[32];
#pragma unroll 1
      for (int m = 0; m < 32; ++m) {
        if (m < Km) {
          int p = i * Km + m, k = i + 1 + m;
          float lv = chart[(((size_t)b * LSEQ + i) * (LSEQ + 1) + k) * NSYM + n];
          float rv = chart[(((size_t)b * LSEQ + k) * (LSEQ + 1) + j) * NSYM + n];
          float beta = mPa[i] - mSa[p];
          bufL[m] = f32_to_bf16(__expf(lv - mLa[p]));
          bufR[m] = f32_to_bf16(__expf(rv - mRa[p] - beta));
        } else { bufL[m] = bf16_zero(); bufR[m] = bf16_zero(); }
      }
      __bf16* dl = eLT + (size_t)row * 32;
      __bf16* dr = eRT + (size_t)row * 32;
#pragma unroll
      for (int q = 0; q < 4; ++q) {
        *(v8bf*)(dl + q * 8) = *(const v8bf*)(bufL + q * 8);
        *(v8bf*)(dr + q * 8) = *(const v8bf*)(bufR + q * 8);
      }
    }
    // zero-fill pad columns of gT (so Phase C loads need no guard)
    {
      unsigned* g32 = (unsigned*)gT;
      for (int idx = cnt * 1152 + tid; idx < 16 * 1152; idx += 256) g32[idx] = 0u;
    }
    __syncthreads();

    // Phase B: G_cc = eL^T @ eR  (48x48, one bf16 WMMA K-tile of 32)
    for (int cc = wave; cc < cnt; cc += 8) {
      for (int tM = 0; tM < 3; ++tM) {
        const __bf16* abase = eLT + ((size_t)cc * NSYM + tM * 16 + r) * 32;
        v8bf va0 = *(const v8bf*)(abase + hi * 8);
        v8bf va1 = *(const v8bf*)(abase + 16 + hi * 8);
        v16bf afrag;
#pragma unroll
        for (int e = 0; e < 8; ++e) { afrag[e] = va0[e]; afrag[e + 8] = va1[e]; }
        for (int tN = 0; tN < 3; ++tN) {
          const __bf16* bbase = eRT + ((size_t)cc * NSYM + tN * 16 + r) * 32;
          v8bf vb0 = *(const v8bf*)(bbase + hi * 8);
          v8bf vb1 = *(const v8bf*)(bbase + 16 + hi * 8);
          v16bf bfrag;
#pragma unroll
          for (int e = 0; e < 8; ++e) { bfrag[e] = vb0[e]; bfrag[e + 8] = vb1[e]; }
          v8f acc = {};
          acc = __builtin_amdgcn_wmma_f32_16x16x32_bf16(false, afrag, false, bfrag,
                                                        (short)0, acc, false, false);
          // D(r,hi)[rr] = G[bL = tM*16+rr+hi*8][bR = tN*16+r]; c = bR*48+bL
          v8bf d;
#pragma unroll
          for (int e = 0; e < 8; ++e) d[e] = f32_to_bf16(acc[e]);
          *(v8bf*)(gT + (size_t)cc * NN + (tN * 16 + r) * NSYM + tM * 16 + hi * 8) = d;
        }
      }
    }
    __syncthreads();

    // Phase C: OUT[a, cc] = W2(48x2304) @ G^T(2304x16); waves 0..5 = (tM, K-half)
    if (wave < 6) {
      const int tM = wave >> 1, kh = wave & 1;
      const int a  = tM * 16 + r;
      const __bf16* wrow = w2 + (size_t)a * NN;
      const __bf16* gcol = gT + (size_t)r * NN;     // lane's column cc = r
      v8f acc = {};
      for (int kt = kh * 36; kt < kh * 36 + 36; ++kt) {
        int kb = kt * 32;
        v8bf va0 = *(const v8bf*)(wrow + kb + hi * 8);
        v8bf va1 = *(const v8bf*)(wrow + kb + 16 + hi * 8);
        v8bf vb0 = *(const v8bf*)(gcol + kb + hi * 8);
        v8bf vb1 = *(const v8bf*)(gcol + kb + 16 + hi * 8);
        v16bf afrag, bfrag;
#pragma unroll
        for (int e = 0; e < 8; ++e) {
          afrag[e] = va0[e]; afrag[e + 8] = va1[e];
          bfrag[e] = vb0[e]; bfrag[e + 8] = vb1[e];
        }
        acc = __builtin_amdgcn_wmma_f32_16x16x32_bf16(false, afrag, false, bfrag,
                                                      (short)0, acc, false, false);
      }
      float* rp = redp + wave * 256;
#pragma unroll
      for (int rr = 0; rr < 8; ++rr) rp[rr * 32 + lane] = acc[rr];
    }
    __syncthreads();

    // combine K-halves, log, write back
    if (wave < 3) {
      const int tM = wave;
      const int icol = i0 + r;
      if (icol < pos) {
        int j = icol + s;
        float addi = mPa[icol];
#pragma unroll
        for (int rr = 0; rr < 8; ++rr) {
          float v = redp[(2 * tM) * 256 + rr * 32 + lane] +
                    redp[(2 * tM + 1) * 256 + rr * 32 + lane];
          int aa = tM * 16 + rr + hi * 8;
          chart[(((size_t)b * LSEQ + icol) * (LSEQ + 1) + j) * NSYM + aa] =
              logf(fmaxf(v, 1e-37f)) + addi + mW[aa];
        }
      }
    }
  }
}

__global__ void final_kernel(const float* __restrict__ chart, float* __restrict__ out) {
  int b = threadIdx.x;
  if (b < 32)
    out[b] = chart[(((size_t)b * LSEQ + 0) * (LSEQ + 1) + LSEQ) * NSYM + 0];
}

extern "C" void kernel_launch(void* const* d_in, const int* in_sizes, int n_in,
                              void* d_out, int out_size, void* d_ws, size_t ws_size,
                              hipStream_t stream) {
  (void)in_sizes; (void)n_in; (void)out_size; (void)ws_size;
  const int B = 32;
  const int* tokens    = (const int*)d_in[0];
  const float* binary  = (const float*)d_in[1];
  const float* lexical = (const float*)d_in[2];
  float* out = (float*)d_out;

  char* ws = (char*)d_ws;
  const size_t CH_BYTES = (size_t)B * LSEQ * (LSEQ + 1) * NSYM * sizeof(float); // 6.49 MB
  float*  chart = (float*)ws;
  __bf16* w2    = (__bf16*)(ws + CH_BYTES);
  float*  mW    = (float*)(ws + CH_BYTES + (size_t)NSYM * NN * sizeof(__bf16));

  (void)hipFuncSetAttribute((const void*)span_step_kernel,
                            hipFuncAttributeMaxDynamicSharedMemorySize, 192 * 1024);

  prep_w_kernel<<<NSYM, 256, 0, stream>>>(binary, w2, mW);
  diag_kernel<<<(B * LSEQ * NSYM + 255) / 256, 256, 0, stream>>>(tokens, lexical, chart);

  for (int s = 2; s <= LSEQ; ++s)
    span_step_kernel<<<B, 256, LDSBYTES, stream>>>(chart, w2, mW, s);

  final_kernel<<<1, 32, 0, stream>>>(chart, out);
}
